// VMamba_Decoder_34626026340768
// MI455X (gfx1250) — compile-verified, matrix operand order checked by reading, not run on hardware
//
#include <hip/hip_runtime.h>
#include <hip/hip_bf16.h>

// ---------------------------------------------------------------------------
// VMamba decoder for MI455X (gfx1250, wave32).
// - Token GEMMs: V_WMMA_F32_16X16X32_F16, f16 operands pre-staged so each
//   fragment is two 16B vector loads; the 16xK weight strip is staged in LDS
//   per block via global_load_async_to_lds_b128 (+ s_wait_asynccnt), shared
//   by 8 waves, B fragments read back as ds_load_b128.
// - Selective scan: 16 SSM states across 16 lanes, __shfl_xor butterflies.
// ---------------------------------------------------------------------------

typedef __attribute__((ext_vector_type(16))) _Float16 v16h;
typedef __attribute__((ext_vector_type(8)))  _Float16 v8h;
typedef __attribute__((ext_vector_type(8)))  float    v8f;
typedef int v4i __attribute__((vector_size(16)));

#define BB   2
#define HH   128
#define WWD  128
#define HWP  (HH*WWD)      // 16384
#define LL   HWP
#define NBLK 4
#define KDIR 4
#define DIN  128
#define NS   16
#define RR   4
#define NTOK (BB*HWP)      // 32768

#if defined(__has_builtin)
#if __has_builtin(__builtin_amdgcn_global_load_async_to_lds_b128) && \
    __has_builtin(__builtin_amdgcn_s_wait_asynccnt)
#define USE_ASYNC_LDS 1
#endif
#endif

__device__ __forceinline__ void stage16(const _Float16* gp, _Float16* lp) {
#if defined(USE_ASYNC_LDS)
  __builtin_amdgcn_global_load_async_to_lds_b128((v4i*)gp, (v4i*)lp, 0, 0);
#else
  *(v8h*)lp = *(const v8h*)gp;
#endif
}
__device__ __forceinline__ void stage_wait() {
#if defined(USE_ASYNC_LDS)
  __builtin_amdgcn_s_wait_asynccnt(0);
#endif
  __syncthreads();
}

__device__ __forceinline__ int kmap16(int lane, int e) {
  int half = (lane >> 4) & 1;
  return (e < 8) ? (half * 8 + e) : (16 + half * 8 + (e - 8));
}

__device__ __forceinline__ v8f wmma_f16(v16h a, v16h b, v8f c) {
  return __builtin_amdgcn_wmma_f32_16x16x32_f16(false, a, false, b, (short)0, c,
                                                false, false);
}

__device__ __forceinline__ v16h frag_cat(v8h lo, v8h hi) {
  return __builtin_shufflevector(lo, hi, 0, 1, 2, 3, 4, 5, 6, 7,
                                 8, 9, 10, 11, 12, 13, 14, 15);
}

// Map scan-sequence index l of direction k to spatial position p = h*W + w.
__device__ __forceinline__ int dirpos(int k, int l) {
  switch (k & 3) {
    case 0:  return l;
    case 1:  return ((l & 127) << 7) | (l >> 7);
    case 2:  return LL - 1 - l;
    default: { int l2 = LL - 1 - l; return ((l2 & 127) << 7) | (l2 >> 7); }
  }
}

// f32 -> f16 conversion (weights, done once per launch).
__global__ __launch_bounds__(256)
void k_cvt(const float* __restrict__ src, _Float16* __restrict__ dst, int n) {
  int i = blockIdx.x * blockDim.x + threadIdx.x;
  if (i < n) dst[i] = (_Float16)src[i];
}

// ---------------------------------------------------------------------------
// f16 token GEMM: C[M,N] (+)= act(A[M,K] * W[N,K]^T + bias).
// Block = one 16-col N-strip x 8 M-tiles (one per wave). W strip staged in
// LDS (async-to-LDS path when available), B fragments via ds_load_b128.
// ---------------------------------------------------------------------------
__global__ __launch_bounds__(256)
void gemm_f16(const _Float16* __restrict__ A, int lda,
              const _Float16* __restrict__ Wt,
              float* __restrict__ Cf, _Float16* __restrict__ Ch, int ldc,
              int M, int N, int Kdim,
              const float* __restrict__ bias, int act, int addC) {
  __shared__ __align__(16) _Float16 wsh[16 * 256];
  const int tilesM8 = (M >> 4) >> 3;
  const int tn = blockIdx.x / tilesM8;
  const int tmBase = (blockIdx.x % tilesM8) << 3;

  const int chunks = (16 * Kdim) >> 3;           // 8-half chunks in the strip
  const _Float16* wsrc = Wt + (size_t)(tn * 16) * Kdim;
  for (int c = threadIdx.x; c < chunks; c += blockDim.x)
    stage16(wsrc + c * 8, wsh + c * 8);
  stage_wait();

  const int lane = threadIdx.x & 31;
  const int wave = threadIdx.x >> 5;
  const int half = lane >> 4;
  const int tm = tmBase + wave;
  const int rowA = tm * 16 + (lane & 15);
  const int rloc = lane & 15;
  const _Float16* arow = A + (size_t)rowA * lda;
  const _Float16* brow = wsh + rloc * Kdim;
  v8f acc = {};
  for (int kb = 0; kb < Kdim; kb += 32) {
    if (kb + 32 < Kdim) __builtin_prefetch(arow + kb + 32, 0, 1);
    v8h alo = *(const v8h*)(arow + kb + half * 8);
    v8h ahi = *(const v8h*)(arow + kb + 16 + half * 8);
    v8h blo = *(const v8h*)(brow + kb + half * 8);
    v8h bhi = *(const v8h*)(brow + kb + 16 + half * 8);
    acc = wmma_f16(frag_cat(alo, ahi), frag_cat(blo, bhi), acc);
  }
  const int col = tn * 16 + (lane & 15);
#pragma unroll
  for (int r = 0; r < 8; ++r) {
    int row = tm * 16 + r + 8 * half;
    float v = acc[r];
    if (bias) v += bias[col];
    if (act == 1) v = 0.5f * v * (1.0f + erff(v * 0.70710678118f));
    size_t idx = (size_t)row * ldc + col;
    if (Ch) { Ch[idx] = (_Float16)v; }
    else    { if (addC) v += Cf[idx]; Cf[idx] = v; }
  }
}

// f = reduce_w (64x128 1x1) applied to concat(x,y); output channel-major.
// (Gather GEMM over f32 inputs; runs once, kept scalar-load form.)
__global__ __launch_bounds__(256)
void k_reduce(const float* __restrict__ x, const float* __restrict__ y,
              const float* __restrict__ rw, float* __restrict__ fout) {
  const int lane = threadIdx.x & 31;
  const int wave = threadIdx.x >> 5;
  const int tile = blockIdx.x * (blockDim.x >> 5) + wave;
  const int tm = tile >> 2, tn = tile & 3;
  const int rowA = tm * 16 + (lane & 15);
  const int b = rowA >> 14, hw = rowA & (HWP - 1);
  const int colW = tn * 16 + (lane & 15);
  v8f acc = {};
  for (int kb = 0; kb < 128; kb += 32) {
    v16h a, bb;
#pragma unroll
    for (int e = 0; e < 16; ++e) {
      int c = kb + kmap16(lane, e);
      float av = (c < 64) ? x[((size_t)(b * 64 + c)) * HWP + hw]
                          : y[((size_t)(b * 64 + (c - 64))) * HWP + hw];
      a[e]  = (_Float16)av;
      bb[e] = (_Float16)rw[colW * 128 + c];
    }
    acc = wmma_f16(a, bb, acc);
  }
  const int half = lane >> 4;
#pragma unroll
  for (int r = 0; r < 8; ++r) {
    int row = tm * 16 + r + 8 * half;
    int ob = row >> 14, ohw = row & (HWP - 1);
    int oc = tn * 16 + (lane & 15);
    fout[((size_t)(ob * 64 + oc)) * HWP + ohw] = acc[r];
  }
}

// t_raw tokens = patch_w (64x64 1x1) applied to f (channel-major input).
__global__ __launch_bounds__(256)
void k_patch(const float* __restrict__ f, const float* __restrict__ pw,
             float* __restrict__ out) {
  const int lane = threadIdx.x & 31;
  const int wave = threadIdx.x >> 5;
  const int tile = blockIdx.x * (blockDim.x >> 5) + wave;
  const int tm = tile >> 2, tn = tile & 3;
  const int rowA = tm * 16 + (lane & 15);
  const int b = rowA >> 14, hw = rowA & (HWP - 1);
  const int colW = tn * 16 + (lane & 15);
  v8f acc = {};
  for (int kb = 0; kb < 64; kb += 32) {
    v16h a, bb;
#pragma unroll
    for (int e = 0; e < 16; ++e) {
      int c = kb + kmap16(lane, e);
      a[e]  = (_Float16)f[((size_t)(b * 64 + c)) * HWP + hw];
      bb[e] = (_Float16)pw[colW * 64 + c];
    }
    acc = wmma_f16(a, bb, acc);
  }
  const int half = lane >> 4;
#pragma unroll
  for (int r = 0; r < 8; ++r) {
    int row = tm * 16 + r + 8 * half;
    out[(size_t)row * 64 + (tn * 16 + (lane & 15))] = acc[r];
  }
}

// LayerNorm over last dim C; writes f32 (outf) or f16 (outh).
__global__ __launch_bounds__(256)
void k_ln(const float* __restrict__ in, float* __restrict__ outf,
          _Float16* __restrict__ outh,
          const float* __restrict__ g, const float* __restrict__ bta,
          int C, int rows) {
  int r = blockIdx.x * blockDim.x + threadIdx.x;
  if (r >= rows) return;
  const float* p = in + (size_t)r * C;
  float s = 0.0f, s2 = 0.0f;
  for (int c = 0; c < C; ++c) { float v = p[c]; s += v; s2 += v * v; }
  float mu  = s / C;
  float var = s2 / C - mu * mu;
  float inv = rsqrtf(var + 1e-5f);
  if (outf) {
    float* o = outf + (size_t)r * C;
    for (int c = 0; c < C; ++c) o[c] = (p[c] - mu) * inv * g[c] + bta[c];
  } else {
    _Float16* o = outh + (size_t)r * C;
    for (int c = 0; c < C; ++c)
      o[c] = (_Float16)((p[c] - mu) * inv * g[c] + bta[c]);
  }
}

// Depthwise 3x3 conv + bias + SiLU. Writes xc32 channel-major (scan u reads)
// and xc16 token-major (x_proj fragment loads).
__global__ __launch_bounds__(256)
void k_dwconv(const float* __restrict__ xz, const float* __restrict__ cw,
              const float* __restrict__ cb, float* __restrict__ xc32,
              _Float16* __restrict__ xc16) {
  int idx = blockIdx.x * blockDim.x + threadIdx.x;
  if (idx >= BB * DIN * HWP) return;
  int hw = idx & (HWP - 1);
  int d  = (idx >> 14) & (DIN - 1);
  int b  = idx >> 21;
  int h = hw >> 7, w = hw & 127;
  float s = cb[d];
  for (int dy = -1; dy <= 1; ++dy) {
    int hh = h + dy; if (hh < 0 || hh >= HH) continue;
    for (int dx = -1; dx <= 1; ++dx) {
      int ww = w + dx; if (ww < 0 || ww >= WWD) continue;
      float v = xz[((size_t)(b * HWP + hh * WWD + ww)) * 256 + d];
      s += v * cw[d * 9 + (dy + 1) * 3 + (dx + 1)];
    }
  }
  float sg = 1.0f / (1.0f + expf(-s));
  float sv = s * sg;
  xc32[idx] = sv;
  xc16[((size_t)(b * HWP + hw)) * DIN + d] = (_Float16)sv;
}

// x_dbl[b,k,c,l] = sum_d x_proj_w[k,c,d] * xc[b,d,dirpos(k,l)].
// Block = (b,k,tn) x 8 M-tiles; 16x128 weight strip staged in LDS with
// zero padding for cols >= 36.
__global__ __launch_bounds__(256)
void k_xproj(const _Float16* __restrict__ xc16, const _Float16* __restrict__ xw,
             float* __restrict__ xdbl) {
  __shared__ __align__(16) _Float16 wsh[16 * DIN];
  int g = blockIdx.x;
  const int tmB = g & 127; g >>= 7;       // 128 groups of 8 M-tiles
  const int tn = g % 3; g /= 3;
  const int b = g >> 2, k = g & 3;

  for (int c = threadIdx.x; c < 16 * (DIN / 8); c += blockDim.x) {
    int row = c >> 4;                     // 16 chunks per row
    int gr = tn * 16 + row;
    _Float16* lp = wsh + c * 8;
    if (gr < 36) {
      stage16(xw + ((size_t)(k * 36 + gr)) * DIN + (c & 15) * 8, lp);
    } else {
      v8h z = {};
      *(v8h*)lp = z;
    }
  }
  stage_wait();

  const int lane = threadIdx.x & 31;
  const int wave = threadIdx.x >> 5;
  const int half = lane >> 4;
  const int tm = tmB * 8 + wave;
  const int lrow = tm * 16 + (lane & 15);
  const int p = dirpos(k, lrow);
  const _Float16* arow = xc16 + ((size_t)(b * HWP + p)) * DIN;
  const _Float16* brow = wsh + (lane & 15) * DIN;
  v8f acc = {};
  for (int kb = 0; kb < DIN; kb += 32) {
    v8h alo = *(const v8h*)(arow + kb + half * 8);
    v8h ahi = *(const v8h*)(arow + kb + 16 + half * 8);
    v8h blo = *(const v8h*)(brow + kb + half * 8);
    v8h bhi = *(const v8h*)(brow + kb + 16 + half * 8);
    acc = wmma_f16(frag_cat(alo, ahi), frag_cat(blo, bhi), acc);
  }
  const int c = tn * 16 + (lane & 15);
  if (c < 36) {
#pragma unroll
    for (int r = 0; r < 8; ++r) {
      int l = tm * 16 + r + 8 * half;
      xdbl[(((size_t)(b * KDIR + k)) * 36 + c) * LL + l] = acc[r];
    }
  }
}

// Selective scan. 16 lanes per (b,k,d) sequence, one SSM state per lane.
__global__ __launch_bounds__(256)
void k_scan(const float* __restrict__ xc, const float* __restrict__ xdbl,
            const float* __restrict__ dtw, const float* __restrict__ dtb,
            const float* __restrict__ alog, const float* __restrict__ dsp,
            float* __restrict__ ys) {
  int gid = blockIdx.x * blockDim.x + threadIdx.x;
  int s = gid >> 4;
  int n = gid & 15;
  if (s >= BB * KDIR * DIN) return;
  int d = s & (DIN - 1);
  int k = (s >> 7) & 3;
  int b = s >> 9;
  float Av = -expf(alog[(k * DIN + d) * NS + n]);
  float Dv = dsp[k * DIN + d];
  float w0 = dtw[(k * DIN + d) * RR + 0];
  float w1 = dtw[(k * DIN + d) * RR + 1];
  float w2 = dtw[(k * DIN + d) * RR + 2];
  float w3 = dtw[(k * DIN + d) * RR + 3];
  float db = dtb[k * DIN + d];
  const float* xd     = xdbl + ((size_t)(b * KDIR + k)) * 36 * LL;
  const float* u_base = xc + ((size_t)(b * DIN + d)) * HWP;
  float* yo = ys + (((size_t)(b * KDIR + k)) * DIN + d) * LL;
  float h = 0.0f;
  for (int l = 0; l < LL; ++l) {
    float dt = w0 * xd[0 * LL + l] + w1 * xd[1 * LL + l] +
               w2 * xd[2 * LL + l] + w3 * xd[3 * LL + l] + db;
    dt = (dt > 20.0f) ? dt : log1pf(expf(dt));
    float u  = u_base[dirpos(k, l)];
    float Bv = xd[(RR + n) * LL + l];
    float Cv = xd[(RR + NS + n) * LL + l];
    h = expf(dt * Av) * h + dt * u * Bv;
    float y = h * Cv;
    y += __shfl_xor(y, 1, 32);
    y += __shfl_xor(y, 2, 32);
    y += __shfl_xor(y, 4, 32);
    y += __shfl_xor(y, 8, 32);
    if (n == 0) yo[l] = y + Dv * u;
  }
}

// Cross-merge 4 directions + out-norm LN + SiLU(z) gating -> f16 yo tokens.
__global__ __launch_bounds__(256)
void k_mergegate(const float* __restrict__ ys, const float* __restrict__ xz,
                 const float* __restrict__ g, const float* __restrict__ bta,
                 _Float16* __restrict__ yo) {
  int tok = blockIdx.x * blockDim.x + threadIdx.x;
  if (tok >= NTOK) return;
  int b = tok >> 14, p = tok & (HWP - 1);
  int h = p >> 7, w = p & 127;
  int lt = w * HH + h;
  const float* y0 = ys + ((size_t)(b * KDIR + 0)) * DIN * LL;
  const float* y1 = ys + ((size_t)(b * KDIR + 1)) * DIN * LL;
  const float* y2 = ys + ((size_t)(b * KDIR + 2)) * DIN * LL;
  const float* y3 = ys + ((size_t)(b * KDIR + 3)) * DIN * LL;
  float s = 0.0f, s2 = 0.0f;
  for (int d = 0; d < DIN; ++d) {
    float v = y0[(size_t)d * LL + p] + y2[(size_t)d * LL + (LL - 1 - p)] +
              y1[(size_t)d * LL + lt] + y3[(size_t)d * LL + (LL - 1 - lt)];
    s += v; s2 += v * v;
  }
  float mu = s / DIN, var = s2 / DIN - mu * mu;
  float inv = rsqrtf(var + 1e-5f);
  for (int d = 0; d < DIN; ++d) {
    float v = y0[(size_t)d * LL + p] + y2[(size_t)d * LL + (LL - 1 - p)] +
              y1[(size_t)d * LL + lt] + y3[(size_t)d * LL + (LL - 1 - lt)];
    v = (v - mu) * inv * g[d] + bta[d];
    float z  = xz[(size_t)tok * 256 + 128 + d];
    float sg = 1.0f / (1.0f + expf(-z));
    yo[(size_t)tok * DIN + d] = (_Float16)(v * (z * sg));
  }
}

// 3x3 conv 64->32 + LeakyReLU(0.01). t is token-major.
__global__ __launch_bounds__(256)
void k_out1(const float* __restrict__ t, const float* __restrict__ w1,
            float* __restrict__ o1) {
  int idx = blockIdx.x * blockDim.x + threadIdx.x;
  if (idx >= BB * 32 * HWP) return;
  int hw = idx & (HWP - 1);
  int o  = (idx >> 14) & 31;
  int b  = idx >> 19;
  int h = hw >> 7, w = hw & 127;
  float s = 0.0f;
  for (int dy = -1; dy <= 1; ++dy) {
    int hh = h + dy; if (hh < 0 || hh >= HH) continue;
    for (int dx = -1; dx <= 1; ++dx) {
      int ww = w + dx; if (ww < 0 || ww >= WWD) continue;
      const float* trow = t + ((size_t)(b * HWP + hh * WWD + ww)) * 64;
      for (int c = 0; c < 64; ++c)
        s += trow[c] * w1[((o * 64 + c) * 3 + (dy + 1)) * 3 + (dx + 1)];
    }
  }
  o1[idx] = (s >= 0.0f) ? s : 0.01f * s;
}

// 3x3 conv 32->1 + inp_img + sigmoid.
__global__ __launch_bounds__(256)
void k_out2(const float* __restrict__ o1, const float* __restrict__ w2,
            const float* __restrict__ img, float* __restrict__ out) {
  int idx = blockIdx.x * blockDim.x + threadIdx.x;
  if (idx >= BB * HWP) return;
  int hw = idx & (HWP - 1);
  int b  = idx >> 14;
  int h = hw >> 7, w = hw & 127;
  float s = 0.0f;
  for (int dy = -1; dy <= 1; ++dy) {
    int hh = h + dy; if (hh < 0 || hh >= HH) continue;
    for (int dx = -1; dx <= 1; ++dx) {
      int ww = w + dx; if (ww < 0 || ww >= WWD) continue;
      for (int c = 0; c < 32; ++c)
        s += o1[((size_t)(b * 32 + c)) * HWP + hh * WWD + ww] *
             w2[(c * 3 + (dy + 1)) * 3 + (dx + 1)];
    }
  }
  s += img[idx];
  out[idx] = 1.0f / (1.0f + expf(-s));
}

extern "C" void kernel_launch(void* const* d_in, const int* in_sizes, int n_in,
                              void* d_out, int out_size, void* d_ws, size_t ws_size,
                              hipStream_t stream) {
  const float* inp_img    = (const float*)d_in[0];
  const float* x          = (const float*)d_in[1];
  const float* y          = (const float*)d_in[2];
  const float* reduce_w   = (const float*)d_in[3];
  const float* patch_w    = (const float*)d_in[4];
  const float* patch_g    = (const float*)d_in[5];
  const float* patch_b    = (const float*)d_in[6];
  const float* ln1_g      = (const float*)d_in[7];
  const float* ln1_b      = (const float*)d_in[8];
  const float* in_proj_w  = (const float*)d_in[9];
  const float* conv_w     = (const float*)d_in[10];
  const float* conv_b     = (const float*)d_in[11];
  const float* x_proj_w   = (const float*)d_in[12];
  const float* dt_proj_w  = (const float*)d_in[13];
  const float* dt_proj_b  = (const float*)d_in[14];
  const float* A_logs     = (const float*)d_in[15];
  const float* Ds_p       = (const float*)d_in[16];
  const float* out_norm_g = (const float*)d_in[17];
  const float* out_norm_b = (const float*)d_in[18];
  const float* out_proj_w = (const float*)d_in[19];
  const float* ln2_g      = (const float*)d_in[20];
  const float* ln2_b      = (const float*)d_in[21];
  const float* fc1_w      = (const float*)d_in[22];
  const float* fc1_b      = (const float*)d_in[23];
  const float* fc2_w      = (const float*)d_in[24];
  const float* fc2_b      = (const float*)d_in[25];
  const float* out1_w     = (const float*)d_in[26];
  const float* out2_w     = (const float*)d_in[27];

  // ---- workspace carve: f32 region, then f16 region ----
  float* fb = (float*)d_ws;
  size_t fo = 0;
  float* t    = fb + fo; fo += (size_t)NTOK * 64;
  float* pbuf = fb + fo; fo += (size_t)NTOK * 64;        // patch GEMM out
  float* xz   = fb + fo; fo += (size_t)NTOK * 256;
  float* xc32 = fb + fo; fo += (size_t)BB * DIN * HWP;   // channel-major
  float* xdbl = fb + fo; fo += (size_t)BB * KDIR * 36 * LL;
  float* ysb  = fb + fo; fo += (size_t)BB * KDIR * DIN * LL;
  float* o1   = fb + fo; fo += (size_t)BB * 32 * HWP;

  _Float16* hb = (_Float16*)(fb + fo);
  size_t ho = 0;
  _Float16* hbuf16 = hb + ho; ho += (size_t)NTOK * 64;
  _Float16* xc16   = hb + ho; ho += (size_t)BB * HWP * DIN;  // token-major
  _Float16* yob16  = hb + ho; ho += (size_t)NTOK * DIN;
  _Float16* w_in   = hb + ho; ho += (size_t)NBLK * 256 * 64;
  _Float16* w_xp   = hb + ho; ho += (size_t)NBLK * KDIR * 36 * DIN;
  _Float16* w_op   = hb + ho; ho += (size_t)NBLK * 64 * 128;
  _Float16* w_f1   = hb + ho; ho += (size_t)NBLK * 256 * 64;
  _Float16* w_f2   = hb + ho; ho += (size_t)NBLK * 64 * 256;
  _Float16* mbuf16 = (_Float16*)xz;  // fc1 out aliases dead xz (f32) region

  float* sigout = (float*)d_out;
  float* fout   = (float*)d_out + BB * HWP;   // second tuple output: f

  dim3 blk(256);

  // Weight conversions (deterministic, every call).
  k_cvt<<<(NBLK*256*64 + 255)/256, blk, 0, stream>>>(in_proj_w, w_in, NBLK*256*64);
  k_cvt<<<(NBLK*KDIR*36*DIN + 255)/256, blk, 0, stream>>>(x_proj_w, w_xp, NBLK*KDIR*36*DIN);
  k_cvt<<<(NBLK*64*128 + 255)/256, blk, 0, stream>>>(out_proj_w, w_op, NBLK*64*128);
  k_cvt<<<(NBLK*256*64 + 255)/256, blk, 0, stream>>>(fc1_w, w_f1, NBLK*256*64);
  k_cvt<<<(NBLK*64*256 + 255)/256, blk, 0, stream>>>(fc2_w, w_f2, NBLK*64*256);

  // Stem: f = reduce(concat(x,y)); t = LN(patch(f))
  k_reduce<<<1024, blk, 0, stream>>>(x, y, reduce_w, fout);
  k_patch <<<1024, blk, 0, stream>>>(fout, patch_w, pbuf);
  k_ln    <<<NTOK / 256, blk, 0, stream>>>(pbuf, t, nullptr, patch_g, patch_b,
                                           64, NTOK);

  for (int i = 0; i < NBLK; ++i) {
    k_ln<<<NTOK / 256, blk, 0, stream>>>(t, nullptr, hbuf16,
                                         ln1_g + i * 64, ln1_b + i * 64, 64, NTOK);
    // in_proj: (32768x64)x(256x64)^T -> xz f32
    gemm_f16<<<16 * 256, blk, 0, stream>>>(hbuf16, 64,
        w_in + (size_t)i * 256 * 64, xz, nullptr, 256, NTOK, 256, 64,
        nullptr, 0, 0);
    k_dwconv<<<(BB * DIN * HWP) / 256, blk, 0, stream>>>(
        xz, conv_w + i * DIN * 9, conv_b + i * DIN, xc32, xc16);
    k_xproj<<<8 * 3 * 128, blk, 0, stream>>>(
        xc16, w_xp + (size_t)i * KDIR * 36 * DIN, xdbl);
    k_scan<<<64, blk, 0, stream>>>(xc32, xdbl,
        dt_proj_w + (size_t)i * KDIR * DIN * RR, dt_proj_b + i * KDIR * DIN,
        A_logs + (size_t)i * KDIR * DIN * NS, Ds_p + i * KDIR * DIN, ysb);
    k_mergegate<<<NTOK / 256, blk, 0, stream>>>(ysb, xz, out_norm_g + i * DIN,
                                                out_norm_b + i * DIN, yob16);
    // out_proj: (32768x128)x(64x128)^T += t
    gemm_f16<<<4 * 256, blk, 0, stream>>>(yob16, 128,
        w_op + (size_t)i * 64 * 128, t, nullptr, 64, NTOK, 64, 128,
        nullptr, 0, 1);
    k_ln<<<NTOK / 256, blk, 0, stream>>>(t, nullptr, hbuf16,
                                         ln2_g + i * 64, ln2_b + i * 64, 64, NTOK);
    // fc1: GELU, f16 out (feeds fc2)
    gemm_f16<<<16 * 256, blk, 0, stream>>>(hbuf16, 64,
        w_f1 + (size_t)i * 256 * 64, nullptr, mbuf16, 256, NTOK, 256, 64,
        fc1_b + i * 256, 1, 0);
    // fc2: += t
    gemm_f16<<<4 * 256, blk, 0, stream>>>(mbuf16, 256,
        w_f2 + (size_t)i * 64 * 256, t, nullptr, 64, NTOK, 64, 256,
        fc2_b + i * 64, 0, 1);
  }

  k_out1<<<(BB * 32 * HWP) / 256, blk, 0, stream>>>(t, out1_w, o1);
  k_out2<<<(BB * HWP) / 256, blk, 0, stream>>>(o1, out2_w, inp_img, sigout);
}
